// Block_24404004176219
// MI455X (gfx1250) — matrix operand order repaired and checked by reading, not compiled, not run on hardware
//
#include <hip/hip_runtime.h>
#include <math.h>

// ---------------------------------------------------------------------------
// Transformer block (B=8, T=2048, C=1024) for MI455X / gfx1250.
// All matmuls via bf16 WMMA (v_wmma_f32_16x16x32_bf16), fp32 accumulate.
// All B operands pre-transposed to [N][K] so LDS staging is pure b128 traffic.
// Double-buffered LDS, software-pipelined (async-to-LDS when available).
// Causal-aware: upper-triangle score blocks skipped, P@V K-loop truncated.
// ---------------------------------------------------------------------------

typedef __attribute__((ext_vector_type(16))) __bf16 bf16x16;
typedef __attribute__((ext_vector_type(8)))  float  f32x8;
typedef __attribute__((ext_vector_type(4)))  int    int4v;
typedef __attribute__((address_space(1))) int4v* gbl_i4p;
typedef __attribute__((address_space(3))) int4v* lds_i4p;

#define BM 128
#define BN 128
#define BK 32
#define LDSB 80          // LDS row pitch in bytes (32 bf16 + 8 pad, 16B aligned)

// epilogue modes
#define EP_SCORES   1    // scale + causal mask, store fp32
#define EP_BF16     2    // store bf16
#define EP_SILU     3    // silu, store bf16
#define EP_RES      4    // add fp32 residual, store fp32

#if __has_builtin(__builtin_amdgcn_global_load_async_to_lds_b128)
#define HAVE_ASYNC_LDS 1
#else
#define HAVE_ASYNC_LDS 0
#endif

__device__ __forceinline__ void wait_asynccnt0() {
#if __has_builtin(__builtin_amdgcn_s_wait_asynccnt)
  __builtin_amdgcn_s_wait_asynccnt(0);
#else
  asm volatile("s_wait_asynccnt 0x0" ::: "memory");
#endif
}

// ---------------------------------------------------------------------------
// fused cast+transpose: fp32 [K][N] -> bf16 [N][K]   (weights, once per call)
// ---------------------------------------------------------------------------
__global__ __launch_bounds__(256) void castT_kernel(
    const float* __restrict__ in, __bf16* __restrict__ out, int K, int N) {
  __shared__ float tile[32][33];
  const int k0 = blockIdx.x * 32, n0 = blockIdx.y * 32;
  const int t = threadIdx.x;
  const int c = t & 31, r = t >> 5;
#pragma unroll
  for (int i = 0; i < 4; ++i)
    tile[r + i * 8][c] = in[(long long)(k0 + r + i * 8) * N + n0 + c];
  __syncthreads();
  const int k = t & 31, n = t >> 5;
#pragma unroll
  for (int i = 0; i < 4; ++i)
    out[(long long)(n0 + n + i * 8) * K + k0 + k] = (__bf16)tile[k][n + i * 8];
}

// ---------------------------------------------------------------------------
// bf16 transpose with leading dims + batch (for V)
// ---------------------------------------------------------------------------
__global__ __launch_bounds__(256) void transT_bf16_kernel(
    const __bf16* __restrict__ in, int ldin, long long strideIn,
    __bf16* __restrict__ out, int ldout, long long strideOut) {
  __shared__ __bf16 tile[32][34];
  const __bf16* inb = in + (long long)blockIdx.z * strideIn;
  __bf16* outb = out + (long long)blockIdx.z * strideOut;
  const int r0 = blockIdx.x * 32, c0 = blockIdx.y * 32;
  const int t = threadIdx.x;
  const int c = t & 31, r = t >> 5;
#pragma unroll
  for (int i = 0; i < 4; ++i)
    tile[r + i * 8][c] = inb[(long long)(r0 + r + i * 8) * ldin + c0 + c];
  __syncthreads();
  const int rr = t & 31, cc = t >> 5;
#pragma unroll
  for (int i = 0; i < 4; ++i)
    outb[(long long)(c0 + cc + i * 8) * ldout + r0 + rr] = tile[rr][cc + i * 8];
}

// ---------------------------------------------------------------------------
// RMSNorm over C=1024, one block (256 thr) per row, bf16 output
// ---------------------------------------------------------------------------
__global__ __launch_bounds__(256) void rmsnorm_kernel(
    const float* __restrict__ x, const float* __restrict__ g,
    __bf16* __restrict__ out, int Cn) {
  long long row = blockIdx.x;
  const float4* xr = (const float4*)(x + row * Cn);
  int t = threadIdx.x;
  float4 v = xr[t];
  float ss = v.x * v.x + v.y * v.y + v.z * v.z + v.w * v.w;
  __shared__ float red[256];
  red[t] = ss;
  __syncthreads();
  for (int s = 128; s > 0; s >>= 1) {
    if (t < s) red[t] += red[t + s];
    __syncthreads();
  }
  float r = rsqrtf(red[0] / (float)Cn + 1e-6f);
  __bf16* orow = out + row * Cn;
  int c = t * 4;
  orow[c + 0] = (__bf16)(v.x * r * g[c + 0]);
  orow[c + 1] = (__bf16)(v.y * r * g[c + 1]);
  orow[c + 2] = (__bf16)(v.z * r * g[c + 2]);
  orow[c + 3] = (__bf16)(v.w * r * g[c + 3]);
}

// ---------------------------------------------------------------------------
// Row softmax over T=2048, one block (256 thr, 8 elems each) per row -> bf16
// ---------------------------------------------------------------------------
__global__ __launch_bounds__(256) void softmax_kernel(
    const float* __restrict__ S, __bf16* __restrict__ P, int T) {
  long long row = blockIdx.x;
  const float* sr = S + row * T;
  int t = threadIdx.x;
  float v[8];
  float mx = -3.0e38f;
#pragma unroll
  for (int i = 0; i < 8; ++i) {
    v[i] = sr[t + i * 256];
    mx = fmaxf(mx, v[i]);
  }
  __shared__ float red[256];
  red[t] = mx;
  __syncthreads();
  for (int s = 128; s > 0; s >>= 1) {
    if (t < s) red[t] = fmaxf(red[t], red[t + s]);
    __syncthreads();
  }
  mx = red[0];
  __syncthreads();
  float sum = 0.f;
#pragma unroll
  for (int i = 0; i < 8; ++i) {
    v[i] = __expf(v[i] - mx);
    sum += v[i];
  }
  red[t] = sum;
  __syncthreads();
  for (int s = 128; s > 0; s >>= 1) {
    if (t < s) red[t] += red[t + s];
    __syncthreads();
  }
  float inv = 1.0f / red[0];
  __bf16* pr = P + row * T;
#pragma unroll
  for (int i = 0; i < 8; ++i) pr[t + i * 256] = (__bf16)(v[i] * inv);
}

// ---------------------------------------------------------------------------
// bf16 WMMA GEMM:  C[b] = A[b] (MxK row-major) * Bt[b]^T  (Bt is [N][K])
// Block tile 128x128, 8 waves (4x2), wave tile 32x64 (2x4 frags, 8 wmma/step).
// Double-buffered LDS; steady-state loop stages next tile unconditionally,
// last K-step peeled.  causalK truncates the K loop at m0+BM (for P@V).
// ---------------------------------------------------------------------------
__global__ __launch_bounds__(256) void gemm_wmma_bf16(
    const __bf16* __restrict__ A, int lda, long long strideA,
    const __bf16* __restrict__ Bt, int ldb, long long strideB,
    float* __restrict__ Cf, __bf16* __restrict__ Cb, int ldc,
    long long strideC, const float* __restrict__ Res, long long strideR,
    int Kdim, int mode, float scale, int causalK) {
  __shared__ __align__(16) char smem[2 * (BM + BN) * LDSB];  // 40960 B
  const int tid = threadIdx.x;
  const int wave = tid >> 5;
  const int lane = tid & 31;
  const int waveM = wave >> 1;   // 0..3 -> 32-row slab
  const int waveN = wave & 1;    // 0..1 -> 64-col slab
  const int half = lane >> 4;
  const int l15 = lane & 15;
  const int m0 = blockIdx.x * BM;
  const int n0 = blockIdx.y * BN;
  const long long zb = blockIdx.z;
  A += zb * strideA;
  Bt += zb * strideB;
  const long long coff = zb * strideC;

  // fully-masked causal score block: fill and exit (no K loop at all)
  if (mode == EP_SCORES && n0 >= m0 + BM) {
#pragma unroll
    for (int mf = 0; mf < 2; ++mf)
#pragma unroll
      for (int nf = 0; nf < 4; ++nf) {
        int n = n0 + waveN * 64 + nf * 16 + l15;
#pragma unroll
        for (int v = 0; v < 8; ++v) {
          int m = m0 + waveM * 32 + mf * 16 + half * 8 + v;
          Cf[coff + (long long)m * ldc + n] = -1.0e30f;
        }
      }
    return;
  }

  const int Keff = causalK ? ((m0 + BM < Kdim) ? (m0 + BM) : Kdim) : Kdim;

  const int eT = tid * 8;  // first element (of 8) this thread stages per half-tile

  auto gldA = [&](int e, int k0) -> uint4 {
    return *(const uint4*)(A + (long long)(m0 + (e >> 5)) * lda + k0 + (e & 31));
  };
  auto gldB = [&](int e, int k0) -> uint4 {
    return *(const uint4*)(Bt + (long long)(n0 + (e >> 5)) * ldb + k0 + (e & 31));
  };
  auto sstA = [&](char* buf, int e, uint4 val) {
    *(uint4*)(buf + (e >> 5) * LDSB + (e & 31) * 2) = val;
  };
  auto sstB = [&](char* buf, int e, uint4 val) {
    *(uint4*)(buf + BM * LDSB + (e >> 5) * LDSB + (e & 31) * 2) = val;
  };

#if HAVE_ASYNC_LDS
  auto asyncStage = [&](char* buf, int k0) {
#pragma unroll
    for (int it = 0; it < 2; ++it) {
      const int e = eT + it * 2048;
      __builtin_amdgcn_global_load_async_to_lds_b128(
          (gbl_i4p)(A + (long long)(m0 + (e >> 5)) * lda + k0 + (e & 31)),
          (lds_i4p)(buf + (e >> 5) * LDSB + (e & 31) * 2),
          0, 0);
      __builtin_amdgcn_global_load_async_to_lds_b128(
          (gbl_i4p)(Bt + (long long)(n0 + (e >> 5)) * ldb + k0 + (e & 31)),
          (lds_i4p)(buf + BM * LDSB + (e >> 5) * LDSB + (e & 31) * 2),
          0, 0);
    }
  };
#endif

  f32x8 acc[2][4] = {};

  auto computeStep = [&](const char* cur) {
    union Frag { bf16x16 v; uint4 q[2]; };
    Frag fa[2];
#pragma unroll
    for (int mf = 0; mf < 2; ++mf) {
      const char* rowA = cur + (waveM * 32 + mf * 16 + l15) * LDSB + half * 16;
      fa[mf].q[0] = *(const uint4*)(rowA);
      fa[mf].q[1] = *(const uint4*)(rowA + 32);
    }
#pragma unroll
    for (int nf = 0; nf < 4; ++nf) {
      Frag fb;
      const char* rowB =
          cur + BM * LDSB + (waveN * 64 + nf * 16 + l15) * LDSB + half * 16;
      fb.q[0] = *(const uint4*)(rowB);
      fb.q[1] = *(const uint4*)(rowB + 32);
      acc[0][nf] = __builtin_amdgcn_wmma_f32_16x16x32_bf16(
          false, fa[0].v, false, fb.v, (short)0, acc[0][nf], false, false);
      acc[1][nf] = __builtin_amdgcn_wmma_f32_16x16x32_bf16(
          false, fa[1].v, false, fb.v, (short)0, acc[1][nf], false, false);
    }
  };

  char* cur = smem;
  char* nxt = smem + (BM + BN) * LDSB;

  // prologue: stage k0=0 into cur
#if HAVE_ASYNC_LDS
  asyncStage(cur, 0);
  wait_asynccnt0();
#else
  {
    uint4 a0 = gldA(eT, 0), a1 = gldA(eT + 2048, 0);
    uint4 b0 = gldB(eT, 0), b1 = gldB(eT + 2048, 0);
    sstA(cur, eT, a0);
    sstA(cur, eT + 2048, a1);
    sstB(cur, eT, b0);
    sstB(cur, eT + 2048, b1);
  }
#endif
  __syncthreads();

  const int nsteps = Keff / BK;
  for (int s = 0; s + 1 < nsteps; ++s) {
    const int kn = (s + 1) * BK;
#if HAVE_ASYNC_LDS
    asyncStage(nxt, kn);          // async into other buffer
    computeStep(cur);             // overlap with WMMA
    wait_asynccnt0();
#else
    uint4 pa0 = gldA(eT, kn), pa1 = gldA(eT + 2048, kn);
    uint4 pb0 = gldB(eT, kn), pb1 = gldB(eT + 2048, kn);
    computeStep(cur);
    sstA(nxt, eT, pa0);
    sstA(nxt, eT + 2048, pa1);
    sstB(nxt, eT, pb0);
    sstB(nxt, eT + 2048, pb1);
#endif
    __syncthreads();
    char* tmp = cur;
    cur = nxt;
    nxt = tmp;
  }
  computeStep(cur);  // peeled last K-step

  // ---- epilogue ----
#pragma unroll
  for (int mf = 0; mf < 2; ++mf) {
#pragma unroll
    for (int nf = 0; nf < 4; ++nf) {
      int n = n0 + waveN * 64 + nf * 16 + l15;
#pragma unroll
      for (int v = 0; v < 8; ++v) {
        int m = m0 + waveM * 32 + mf * 16 + half * 8 + v;
        float val = acc[mf][nf][v];
        long long o = coff + (long long)m * ldc + n;
        if (mode == EP_SCORES) {
          val *= scale;
          if (n > m) val = -1.0e30f;  // causal mask
          Cf[o] = val;
        } else if (mode == EP_BF16) {
          Cb[o] = (__bf16)val;
        } else if (mode == EP_SILU) {
          Cb[o] = (__bf16)(val / (1.0f + __expf(-val)));
        } else {  // EP_RES
          Cf[o] = Res[zb * strideR + (long long)m * ldc + n] + val;
        }
      }
    }
  }
}

// ---------------------------------------------------------------------------
// Host-side orchestration
// ---------------------------------------------------------------------------
extern "C" void kernel_launch(void* const* d_in, const int* in_sizes, int n_in,
                              void* d_out, int out_size, void* d_ws,
                              size_t ws_size, hipStream_t stream) {
  (void)in_sizes; (void)n_in; (void)out_size; (void)ws_size;
  constexpr int Cdim = 1024, Tdim = 2048, Bdim = 8;
  constexpr long long BT = (long long)Bdim * Tdim;  // 16384

  const float* x      = (const float*)d_in[0];
  const float* w_qkv  = (const float*)d_in[1];
  const float* w_proj = (const float*)d_in[2];
  const float* w1     = (const float*)d_in[3];
  const float* w2     = (const float*)d_in[4];
  const float* g1     = (const float*)d_in[5];
  const float* g2     = (const float*)d_in[6];
  float* out = (float*)d_out;

  // workspace carve (256B aligned)
  char* ws = (char*)d_ws;
  auto alloc = [&](size_t bytes) -> char* {
    char* p = ws;
    ws += (bytes + 255) & ~(size_t)255;
    return p;
  };
  __bf16* wqkvT  = (__bf16*)alloc((size_t)3 * Cdim * Cdim * 2);  // [3C][C]
  __bf16* wprojT = (__bf16*)alloc((size_t)Cdim * Cdim * 2);      // [C][C]
  __bf16* w1T    = (__bf16*)alloc((size_t)2 * Cdim * Cdim * 2);  // [2C][C]
  __bf16* w2T    = (__bf16*)alloc((size_t)Cdim * 2 * Cdim * 2);  // [C][2C]
  __bf16* h_bf   = (__bf16*)alloc((size_t)BT * Cdim * 2);
  __bf16* qkv_bf = (__bf16*)alloc((size_t)BT * 3 * Cdim * 2);
  __bf16* vT     = (__bf16*)alloc((size_t)BT * Cdim * 2);        // [B][C][T]
  float*  scores = (float*)alloc((size_t)Bdim * Tdim * Tdim * 4);
  __bf16* P_bf   = (__bf16*)alloc((size_t)Bdim * Tdim * Tdim * 2);
  __bf16* ao_bf  = (__bf16*)alloc((size_t)BT * Cdim * 2);
  float*  x2     = (float*)alloc((size_t)BT * Cdim * 4);
  __bf16* h2_bf  = (__bf16*)alloc((size_t)BT * Cdim * 2);
  __bf16* h1_bf  = (__bf16*)alloc((size_t)BT * 2 * Cdim * 2);

  // --- weight cast+transpose: fp32 [K][N] -> bf16 [N][K] ---
  castT_kernel<<<dim3(Cdim / 32, 3 * Cdim / 32), 256, 0, stream>>>(
      w_qkv, wqkvT, Cdim, 3 * Cdim);
  castT_kernel<<<dim3(Cdim / 32, Cdim / 32), 256, 0, stream>>>(
      w_proj, wprojT, Cdim, Cdim);
  castT_kernel<<<dim3(Cdim / 32, 2 * Cdim / 32), 256, 0, stream>>>(
      w1, w1T, Cdim, 2 * Cdim);
  castT_kernel<<<dim3(2 * Cdim / 32, Cdim / 32), 256, 0, stream>>>(
      w2, w2T, 2 * Cdim, Cdim);

  // --- rmsnorm 1 ---
  rmsnorm_kernel<<<(int)BT, 256, 0, stream>>>(x, g1, h_bf, Cdim);

  // --- qkv = h @ w_qkv  (bf16 out) ---
  gemm_wmma_bf16<<<dim3((int)(BT / BM), 3 * Cdim / BN, 1), 256, 0, stream>>>(
      h_bf, Cdim, 0, wqkvT, Cdim, 0,
      nullptr, qkv_bf, 3 * Cdim, 0, nullptr, 0, Cdim, EP_BF16, 1.0f, 0);

  // --- scores = q @ k^T * C^-0.5, causal mask (per batch) ---
  gemm_wmma_bf16<<<dim3(Tdim / BM, Tdim / BN, Bdim), 256, 0, stream>>>(
      qkv_bf + 0, 3 * Cdim, (long long)Tdim * 3 * Cdim,
      qkv_bf + Cdim, 3 * Cdim, (long long)Tdim * 3 * Cdim,
      scores, nullptr, Tdim, (long long)Tdim * Tdim,
      nullptr, 0, Cdim, EP_SCORES, 0.03125f /* 1/sqrt(1024) */, 0);

  // --- softmax rows -> bf16 P ---
  softmax_kernel<<<(int)(Bdim * (long long)Tdim), 256, 0, stream>>>(
      scores, P_bf, Tdim);

  // --- transpose V: [T][C] slice of qkv -> vT [C][T] per batch ---
  transT_bf16_kernel<<<dim3(Tdim / 32, Cdim / 32, Bdim), 256, 0, stream>>>(
      qkv_bf + 2 * Cdim, 3 * Cdim, (long long)Tdim * 3 * Cdim,
      vT, Tdim, (long long)Cdim * Tdim);

  // --- attn_out = P @ V  (bf16 out, per batch, K truncated causally) ---
  gemm_wmma_bf16<<<dim3(Tdim / BM, Cdim / BN, Bdim), 256, 0, stream>>>(
      P_bf, Tdim, (long long)Tdim * Tdim,
      vT, Tdim, (long long)Cdim * Tdim,
      nullptr, ao_bf, Cdim, (long long)Tdim * Cdim,
      nullptr, 0, Tdim, EP_BF16, 1.0f, 1);

  // --- x2 = x + attn_out @ w_proj ---
  gemm_wmma_bf16<<<dim3((int)(BT / BM), Cdim / BN, 1), 256, 0, stream>>>(
      ao_bf, Cdim, 0, wprojT, Cdim, 0,
      x2, nullptr, Cdim, 0, x, 0, Cdim, EP_RES, 1.0f, 0);

  // --- rmsnorm 2 ---
  rmsnorm_kernel<<<(int)BT, 256, 0, stream>>>(x2, g2, h2_bf, Cdim);

  // --- h1 = silu(h2 @ w1)  (bf16 out) ---
  gemm_wmma_bf16<<<dim3((int)(BT / BM), 2 * Cdim / BN, 1), 256, 0, stream>>>(
      h2_bf, Cdim, 0, w1T, Cdim, 0,
      nullptr, h1_bf, 2 * Cdim, 0, nullptr, 0, Cdim, EP_SILU, 1.0f, 0);

  // --- out = x2 + h1 @ w2 ---
  gemm_wmma_bf16<<<dim3((int)(BT / BM), Cdim / BN, 1), 256, 0, stream>>>(
      h1_bf, 2 * Cdim, 0, w2T, 2 * Cdim, 0,
      out, nullptr, Cdim, 0, x2, 0, 2 * Cdim, EP_RES, 1.0f, 0);
}
